// ComplexLSTM_45518063403239
// MI455X (gfx1250) — matrix operand order repaired and checked by reading, not compiled, not run on hardware
//
#include <hip/hip_runtime.h>
#include <hip/hip_bf16.h>

typedef __attribute__((ext_vector_type(16))) _Float16 v16h;
typedef __attribute__((ext_vector_type(8)))  float    v8f;

// Problem dims: B=64, I=128, T=1024, H=256, 4H=1024

// ---------------- workspace layout (bytes) ----------------
#define OFF_XPACK   (0u)
#define SZ_XPACK    (2u*1024u*4u*4u*512u*2u)     /* 33,554,432: [part][t][mt][kt][lane][slot] f16 */
#define OFF_BIH     (OFF_XPACK + SZ_XPACK)
#define SZ_BIH      (2u*64u*4u*512u*2u)          /* 524,288:  [set][ntile][kt][lane][slot] f16 */
#define OFF_BHH     (OFF_BIH + SZ_BIH)
#define SZ_BHH      (2u*64u*8u*512u*2u)          /* 1,048,576 */
#define OFF_BIAS    (OFF_BHH + SZ_BHH)
#define SZ_BIAS     (2u*1024u*4u)
#define OFF_HOUT    (OFF_BIAS + SZ_BIAS)
#define SZ_HOUT     (4u*64u*1024u*256u*2u)       /* 134,217,728: [run][b][t][h] f16 */

// A-fragment (16x32 f16) lane/slot map per CDNA5 ISA 7.12.2:
// lane l(0..31): m = l&15 ; slot s(0..15) in v16h (VGPR g = s>>1, half = s&1)
// k_local = (g>=4?16:0) + (l>=16?8:0) + 2*(g&3) + half
__device__ __forceinline__ void a_slot(int m, int klocal, int& lane, int& slot) {
    lane = m + (((klocal & 15) >= 8) ? 16 : 0);
    slot = 2 * (((klocal >= 16) ? 4 : 0) + ((klocal & 7) >> 1)) + (klocal & 1);
}

// Hardware tanh if available (CDNA5 trans op V_TANH_F32), else exp-based fallback.
#if __has_builtin(__builtin_amdgcn_tanhf)
__device__ __forceinline__ float htanh(float x) { return __builtin_amdgcn_tanhf(x); }
#elif __has_builtin(__builtin_amdgcn_tanh_f32)
__device__ __forceinline__ float htanh(float x) { return __builtin_amdgcn_tanh_f32(x); }
#else
__device__ __forceinline__ float htanh(float x) { return 2.0f / (1.0f + __expf(-2.0f * x)) - 1.0f; }
#endif
__device__ __forceinline__ float hsig(float x) { return 0.5f * htanh(0.5f * x) + 0.5f; }

// ---------------- prepass: pack x into A-fragment order ----------------
// x: [2,B,I,T] f32  ->  xpack: [part][t][mt][kt][lane][slot] f16
__global__ void pack_x(const float* __restrict__ x, _Float16* __restrict__ xpack) {
    unsigned tid = blockIdx.x * blockDim.x + threadIdx.x;  // 16,777,216
    int t    = tid & 1023;
    int k    = (tid >> 10) & 127;
    int b    = (tid >> 17) & 63;
    int part = tid >> 23;
    float v = x[tid];                       // source layout matches decomposition (t fastest)
    int mt = b >> 4, m = b & 15;
    int kt = k >> 5, kl = k & 31;
    int lane, slot; a_slot(m, kl, lane, slot);
    size_t di = ((((size_t)part * 1024 + t) * 4 + mt) * 4 + kt) * 512 + (size_t)lane * 16 + slot;
    xpack[di] = (_Float16)v;
}

// Whh [4H,H] -> B-fragments of Whh^T: lane = (n&15) + ((kl&15)>=8?16:0), slot = (kl>=16?8:0) + (kl&7)
__global__ void pack_whh(const float* __restrict__ Wr, const float* __restrict__ Wi,
                         _Float16* __restrict__ dst) {
    unsigned tid = blockIdx.x * blockDim.x + threadIdx.x;  // 524,288
    int k = tid & 255;
    int n = (tid >> 8) & 1023;
    int set = tid >> 18;
    float v = (set ? Wi : Wr)[n * 256 + k];
    int ntile = n >> 4, nl = n & 15;
    int kt = k >> 5, kl = k & 31;
    int lane = nl + (((kl & 15) >= 8) ? 16 : 0);
    int slot = ((kl >= 16) ? 8 : 0) + (kl & 7);
    size_t di = (((size_t)set * 64 + ntile) * 8 + kt) * 512 + (size_t)lane * 16 + slot;
    dst[di] = (_Float16)v;
}

__global__ void pack_wih(const float* __restrict__ Wr, const float* __restrict__ Wi,
                         _Float16* __restrict__ dst) {
    unsigned tid = blockIdx.x * blockDim.x + threadIdx.x;  // 262,144
    int k = tid & 127;
    int n = (tid >> 7) & 1023;
    int set = tid >> 17;
    float v = (set ? Wi : Wr)[n * 128 + k];
    int ntile = n >> 4, nl = n & 15;
    int kt = k >> 5, kl = k & 31;
    int lane = nl + (((kl & 15) >= 8) ? 16 : 0);
    int slot = ((kl >= 16) ? 8 : 0) + (kl & 7);
    size_t di = (((size_t)set * 64 + ntile) * 4 + kt) * 512 + (size_t)lane * 16 + slot;
    dst[di] = (_Float16)v;
}

__global__ void pack_bias(const float* __restrict__ br1, const float* __restrict__ br2,
                          const float* __restrict__ bi1, const float* __restrict__ bi2,
                          float* __restrict__ bias) {
    unsigned tid = blockIdx.x * blockDim.x + threadIdx.x;  // 2048
    int n = tid & 1023;
    int set = tid >> 10;
    bias[tid] = set ? (bi1[n] + bi2[n]) : (br1[n] + br2[n]);
}

// ---------------- persistent recurrent kernel ----------------
// grid = 4 blocks (one per LSTM run), block = 512 threads (16 waves), M = 64 (whole batch).
// Wave w owns h-columns [w*16, w*16+16): n-tiles {w, w+16, w+32, w+48} = i/f/g/o of the same
// columns for all 4 M-tiles, so gate math is register-local; c-state = 32 VGPRs/lane.
// Weight B-fragments stream from L2 every step; an opaque 32-bit offset (added to the
// provenance-preserving global pointers) defeats LICM without degrading to FLAT loads.
// Accumulators start at inline C=0; biases are added in the epilogue (no splat registers).
__global__ void __launch_bounds__(512, 1)
lstm_recurrent(const _Float16* __restrict__ xpack,
               const _Float16* __restrict__ bihp,
               const _Float16* __restrict__ bhhp,
               const float*    __restrict__ bias,
               _Float16*       __restrict__ hout) {
    __shared__ __align__(32) _Float16 xlds[2][16][32][16];     // [buf][mt*4+kt][lane][slot] 32KB
    __shared__ __align__(32) _Float16 hlds[2][4][8][32][16];   // [buf][mt][kt][lane][slot]  64KB

    const int tid   = threadIdx.x;
    const int lane  = tid & 31;
    const int w     = tid >> 5;          // wave 0..15
    const int run   = blockIdx.x;        // 0:r(xr) 1:i(xi) 2:r(xi) 3:i(xr)
    const int wset  = run & 1;           // 0 -> W_r, 1 -> W_i
    const int xpart = (run == 1 || run == 2) ? 1 : 0;
    const int fstage = tid >> 5;         // staged x fragment id (= wave id)
    const int hi8   = (lane >= 16) ? 8 : 0;

    const v16h* xp_v = (const v16h*)xpack;

    // ---- prologue: zero h buffer 0, stage x_0 into xlds[0] ----
    for (int idx = tid; idx < 4 * 8 * 32 * 16; idx += 512)
        ((_Float16*)hlds)[idx] = (_Float16)0.0f;
    *(v16h*)&xlds[0][fstage][lane][0] =
        xp_v[(((size_t)xpart * 1024 + 0) * 16 + fstage) * 32 + lane];
    __syncthreads();

    float bias_v[4];
#pragma unroll
    for (int G = 0; G < 4; ++G)
        bias_v[G] = bias[wset * 1024 + G * 256 + w * 16 + (lane & 15)];

    float cst[4][8];
#pragma unroll
    for (int mt = 0; mt < 4; ++mt)
#pragma unroll
        for (int v = 0; v < 8; ++v) cst[mt][v] = 0.0f;

    // h-write swizzle (constant per wave/lane)
    const int jcol = w * 16 + (lane & 15);
    const int kt_h = jcol >> 5;
    const int kl   = jcol & 31;
    const int lt0  = (((kl & 15) >= 8) ? 16 : 0);
    const int slot = 2 * (((kl >= 16) ? 4 : 0) + ((kl & 7) >> 1)) + (kl & 1);

    // per-wave weight fragment bases; (G,kt) offsets are immediates (< 8MB)
    const _Float16* pih = bihp + ((size_t)(wset * 64 + w) * 4) * 512 + (size_t)lane * 16;
    const _Float16* phh = bhhp + ((size_t)(wset * 64 + w) * 8) * 512 + (size_t)lane * 16;

    // per-m-tile output bases; (v,t) offsets are immediates / one t*256 term
    _Float16* hb[4];
#pragma unroll
    for (int mt = 0; mt < 4; ++mt)
        hb[mt] = hout + (((size_t)run * 64 + mt * 16 + hi8) * 1024) * 256 + jcol;

    for (int t = 0; t < 1024; ++t) {
        const int cur = t & 1, nxt = cur ^ 1;

        // ---- cooperative stage of x_{t+1} (16 KB) into the other buffer ----
        {
            int tt = (t + 1) & 1023;
            *(v16h*)&xlds[nxt][fstage][lane][0] =
                xp_v[(((size_t)xpart * 1024 + tt) * 16 + fstage) * 32 + lane];
        }

        // opaque (formally loop-variant) 32-bit element offset: defeats LICM of the 48
        // weight fragments while keeping global address-space provenance (global_load).
        unsigned zoff = 0;
        asm volatile("" : "+v"(zoff));
        const _Float16* qih = pih + zoff;
        const _Float16* qhh = phh + zoff;

        v8f acc[4][4];                      // [gate][mtile]; zero -> inline C=0 on 1st WMMA
#pragma unroll
        for (int G = 0; G < 4; ++G)
#pragma unroll
            for (int mt = 0; mt < 4; ++mt)
#pragma unroll
                for (int v = 0; v < 8; ++v) acc[G][mt][v] = 0.0f;

        // ---- gates += x_t*Wih^T + h_t*Whh^T : 12 k-tiles, per k-tile load all 4 gate
        //      fragments in one clause, then 16 WMMAs with staggered loadcnt waits ----
#pragma unroll
        for (int kt = 0; kt < 12; ++kt) {
            v16h bf0, bf1, bf2, bf3;
            if (kt < 4) {
                bf0 = *(const v16h*)(qih + (0 * 64 + kt) * 512);
                bf1 = *(const v16h*)(qih + (1 * 64 + kt) * 512);
                bf2 = *(const v16h*)(qih + (2 * 64 + kt) * 512);
                bf3 = *(const v16h*)(qih + (3 * 64 + kt) * 512);
            } else {
                bf0 = *(const v16h*)(qhh + (0 * 128 + (kt - 4)) * 512);
                bf1 = *(const v16h*)(qhh + (1 * 128 + (kt - 4)) * 512);
                bf2 = *(const v16h*)(qhh + (2 * 128 + (kt - 4)) * 512);
                bf3 = *(const v16h*)(qhh + (3 * 128 + (kt - 4)) * 512);
            }
            v16h a0, a1, a2, a3;
            if (kt < 4) {
                a0 = *(const v16h*)&xlds[cur][0 * 4 + kt][lane][0];
                a1 = *(const v16h*)&xlds[cur][1 * 4 + kt][lane][0];
                a2 = *(const v16h*)&xlds[cur][2 * 4 + kt][lane][0];
                a3 = *(const v16h*)&xlds[cur][3 * 4 + kt][lane][0];
            } else {
                a0 = *(const v16h*)&hlds[cur][0][kt - 4][lane][0];
                a1 = *(const v16h*)&hlds[cur][1][kt - 4][lane][0];
                a2 = *(const v16h*)&hlds[cur][2][kt - 4][lane][0];
                a3 = *(const v16h*)&hlds[cur][3][kt - 4][lane][0];
            }
#pragma unroll
            for (int G = 0; G < 4; ++G) {
                v16h bf = (G == 0) ? bf0 : (G == 1) ? bf1 : (G == 2) ? bf2 : bf3;
                acc[G][0] = __builtin_amdgcn_wmma_f32_16x16x32_f16(
                    false, a0, false, bf, (short)0, acc[G][0], false, false);
                acc[G][1] = __builtin_amdgcn_wmma_f32_16x16x32_f16(
                    false, a1, false, bf, (short)0, acc[G][1], false, false);
                acc[G][2] = __builtin_amdgcn_wmma_f32_16x16x32_f16(
                    false, a2, false, bf, (short)0, acc[G][2], false, false);
                acc[G][3] = __builtin_amdgcn_wmma_f32_16x16x32_f16(
                    false, a3, false, bf, (short)0, acc[G][3], false, false);
            }
        }

        // ---- register-local gate math (bias added here); update c, emit h ----
#pragma unroll
        for (int mt = 0; mt < 4; ++mt) {
#pragma unroll
            for (int v = 0; v < 8; ++v) {
                const int ml = v + hi8;                    // local row in m-tile
                float iv = hsig(acc[0][mt][v] + bias_v[0]);
                float fv = hsig(acc[1][mt][v] + bias_v[1]);
                float gv = htanh(acc[2][mt][v] + bias_v[2]);
                float ov = hsig(acc[3][mt][v] + bias_v[3]);
                float c  = fv * cst[mt][v] + iv * gv;
                cst[mt][v] = c;
                float h  = ov * htanh(c);
                hlds[nxt][mt][kt_h][lt0 + ml][slot] = (_Float16)h;
                hb[mt][(size_t)t * 256 + (size_t)v * 262144] = (_Float16)h;
            }
        }
        __syncthreads();   // next-step buffers (x and h) complete; WAR on cur buffers safe
    }
}

// ---------------- combine: out[2,B,H,T] = (run0 - run1, run2 + run3) ----------------
// LDS-tiled transpose: coalesced reads along hc, coalesced writes along t.
// grid: part(2) x b(64) x hcblk(8 of 32) x tblk(4 of 256); block = 256 threads.
__global__ void combine(const _Float16* __restrict__ hout, float* __restrict__ out) {
    __shared__ float tile[32][257];   // 257-float pitch: conflict-free both phases

    unsigned bid = blockIdx.x;                 // 4096 blocks
    int tblk  = bid & 3;
    int hcblk = (bid >> 2) & 7;
    int b     = (bid >> 5) & 63;
    int part  = bid >> 11;
    int r0 = part * 2, r1 = part * 2 + 1;
    int tid = threadIdx.x;

    // phase 1: read [t][hc] coalesced (hc fastest), combine, store transposed
#pragma unroll
    for (int p = 0; p < 32; ++p) {
        int idx = p * 256 + tid;               // 8192 elements = 256 t x 32 hc
        int hc2 = idx & 31;
        int t2  = idx >> 5;
        size_t ia = (((size_t)r0 * 64 + b) * 1024 + tblk * 256 + t2) * 256 + hcblk * 32 + hc2;
        size_t ib = (((size_t)r1 * 64 + b) * 1024 + tblk * 256 + t2) * 256 + hcblk * 32 + hc2;
        float va = (float)hout[ia];
        float vb = (float)hout[ib];
        tile[hc2][t2] = part ? (va + vb) : (va - vb);
    }
    __syncthreads();

    // phase 2: write [hc][t] coalesced (t fastest)
#pragma unroll
    for (int p = 0; p < 32; ++p) {
        int idx = p * 256 + tid;
        int t2  = idx & 255;
        int hc2 = idx >> 8;
        out[(((size_t)part * 64 + b) * 256 + hcblk * 32 + hc2) * 1024 + tblk * 256 + t2] =
            tile[hc2][t2];
    }
}

extern "C" void kernel_launch(void* const* d_in, const int* in_sizes, int n_in,
                              void* d_out, int out_size, void* d_ws, size_t ws_size,
                              hipStream_t stream) {
    const float* x     = (const float*)d_in[0];
    const float* Wih_r = (const float*)d_in[1];
    const float* Whh_r = (const float*)d_in[2];
    const float* bih_r = (const float*)d_in[3];
    const float* bhh_r = (const float*)d_in[4];
    const float* Wih_i = (const float*)d_in[5];
    const float* Whh_i = (const float*)d_in[6];
    const float* bih_i = (const float*)d_in[7];
    const float* bhh_i = (const float*)d_in[8];

    char* ws = (char*)d_ws;
    _Float16* xpack = (_Float16*)(ws + OFF_XPACK);
    _Float16* bihp  = (_Float16*)(ws + OFF_BIH);
    _Float16* bhhp  = (_Float16*)(ws + OFF_BHH);
    float*    biasp = (float*)   (ws + OFF_BIAS);
    _Float16* houtp = (_Float16*)(ws + OFF_HOUT);
    float*    out   = (float*)d_out;

    pack_x   <<<16777216 / 256, 256, 0, stream>>>(x, xpack);
    pack_whh <<<  524288 / 256, 256, 0, stream>>>(Whh_r, Whh_i, bhhp);
    pack_wih <<<  262144 / 256, 256, 0, stream>>>(Wih_r, Wih_i, bihp);
    pack_bias<<<       8,       256, 0, stream>>>(bih_r, bhh_r, bih_i, bhh_i, biasp);

    lstm_recurrent<<<4, 512, 0, stream>>>(xpack, bihp, bhhp, biasp, houtp);

    combine<<<4096, 256, 0, stream>>>(houtp, out);
}
